// SwinLayer_26663156973985
// MI455X (gfx1250) — matrix operand order repaired and checked by reading, not compiled, not run on hardware
//
#include <hip/hip_runtime.h>
#include <hip/hip_bf16.h>

// ---------- types ----------
typedef __attribute__((ext_vector_type(16))) __bf16 v16bf;
typedef __attribute__((ext_vector_type(8)))  float  v8f;
typedef __attribute__((ext_vector_type(4)))  unsigned int u32x4;

union Frag { v16bf v; u32x4 q[2]; unsigned short h[16]; };

#define CDIM 384
#define NTOK 100352   // 32 * 56 * 56

__device__ __forceinline__ unsigned short f2bf(float f) {
  union { float f; unsigned u; } v; v.f = f;
  unsigned r = v.u + 0x7FFFu + ((v.u >> 16) & 1u);   // RNE
  return (unsigned short)(r >> 16);
}

// window-ordered row -> token index (cyclic shift by +3 both axes)
__device__ __forceinline__ int winmap(int r) {
  int bidx = r / 3136;
  int rem  = r - bidx * 3136;
  int wseq = rem / 49;
  int l    = rem - wseq * 49;
  int wh = wseq >> 3, ww = wseq & 7;
  int h = wh * 7 + l / 7;
  int w = ww * 7 + l % 7;
  h += 3; if (h >= 56) h -= 56;
  w += 3; if (w >= 56) w -= 56;
  return bidx * 3136 + h * 56 + w;
}

// ---------- weight conversion ----------
__global__ void f32_to_bf16(const float* __restrict__ in,
                            unsigned short* __restrict__ out, int n) {
  int i = blockIdx.x * 256 + threadIdx.x;
  if (i < n) out[i] = f2bf(in[i]);
}

// ---------- precompute additive attention term: [w64][head][64][64] ----------
// value = rel_bias(h,m,n) + shift_mask(w64,m,n) for m,n < 49, else -1e30
__global__ void build_biasmask(const float* __restrict__ btab,  // (169 x 12)
                               float* __restrict__ bm) {
  int idx = blockIdx.x * 256 + threadIdx.x;
  if (idx >= 64 * 12 * 64 * 64) return;
  int n   = idx & 63;
  int m   = (idx >> 6) & 63;
  int h   = (idx >> 12) % 12;
  int w64 = idx / (12 * 64 * 64);
  float val = -1e30f;
  if (m < 49 && n < 49) {
    int mq0 = m / 7, mq1 = m % 7, nk0 = n / 7, nk1 = n % 7;
    int ridx = (mq0 - nk0 + 6) * 13 + (mq1 - nk1 + 6);
    val = btab[ridx * 12 + h];
    int wh = w64 >> 3, ww = w64 & 7;
    int hm = wh * 7 + mq0, wm = ww * 7 + mq1;
    int hn = wh * 7 + nk0, wn = ww * 7 + nk1;
    int lm  = (hm < 49 ? 0 : (hm < 53 ? 1 : 2)) * 3 + (wm < 49 ? 0 : (wm < 53 ? 1 : 2));
    int lnn = (hn < 49 ? 0 : (hn < 53 ? 1 : 2)) * 3 + (wn < 49 ? 0 : (wn < 53 ? 1 : 2));
    if (lm != lnn) val -= 100.0f;
  }
  bm[idx] = val;
}

// ---------- layernorm (optionally gathering through the shift/window map) ----------
template<bool GATHER>
__global__ __launch_bounds__(128)
void layernorm_kernel(const float* __restrict__ in, const float* __restrict__ g,
                      const float* __restrict__ b, unsigned short* __restrict__ out) {
  int r = blockIdx.x;
  int src = GATHER ? winmap(r) : r;
  const float* x = in + (size_t)src * CDIM;
  float vals[3]; float s = 0.f, ss = 0.f;
  #pragma unroll
  for (int j = 0; j < 3; ++j) {
    float v = x[threadIdx.x + 128 * j];
    vals[j] = v; s += v; ss += v * v;
  }
  #pragma unroll
  for (int off = 16; off > 0; off >>= 1) {
    s  += __shfl_xor(s,  off, 32);
    ss += __shfl_xor(ss, off, 32);
  }
  __shared__ float red[8];
  int wid = threadIdx.x >> 5;
  if ((threadIdx.x & 31) == 0) { red[wid] = s; red[4 + wid] = ss; }
  __syncthreads();
  s  = red[0] + red[1] + red[2] + red[3];
  ss = red[4] + red[5] + red[6] + red[7];
  float mu  = s * (1.f / 384.f);
  float var = ss * (1.f / 384.f) - mu * mu;
  float rs  = rsqrtf(var + 1e-5f);
  unsigned short* o = out + (size_t)r * CDIM;
  #pragma unroll
  for (int j = 0; j < 3; ++j) {
    int c = threadIdx.x + 128 * j;
    o[c] = f2bf((vals[j] - mu) * rs * g[c] + b[c]);
  }
}

// ---------- generic bf16 WMMA GEMM: out = A(MxK) @ W(NxK)^T + bias ----------
// MODE 0: bf16 store (QKV)             MODE 1: winmap scatter + residual, f32 (out-proj)
// MODE 2: GELU + bf16 store (FC1)      MODE 3: residual + f32 store (FC2 -> d_out)
template<int MODE, int KV>
__global__ __launch_bounds__(256)
void gemm_wmma(const unsigned short* __restrict__ A,
               const unsigned short* __restrict__ W,
               const float* __restrict__ bias,
               void* __restrict__ outp,
               const float* __restrict__ resid,
               int N) {
  const int tilesN = N >> 4;
  int tile  = blockIdx.x * 8 + (threadIdx.x >> 5);
  int tileM = tile / tilesN;
  int tileN = tile - tileM * tilesN;
  int lane = threadIdx.x & 31;
  int l16  = lane & 15;
  int half = lane >> 4;

  const unsigned short* arow = A + (size_t)(tileM * 16 + l16) * KV;
  const unsigned short* wrow = W + (size_t)(tileN * 16 + l16) * KV;
  v8f acc = {};
  #pragma unroll 4
  for (int k0 = 0; k0 < KV; k0 += 32) {
    Frag a, b;
    a.q[0] = *(const u32x4*)(arow + k0 + half * 8);       // K: base..base+7
    a.q[1] = *(const u32x4*)(arow + k0 + half * 8 + 16);  // K: base+16..base+23
    b.q[0] = *(const u32x4*)(wrow + k0 + half * 8);       // K: base..base+7
    b.q[1] = *(const u32x4*)(wrow + k0 + half * 8 + 16);  // K: base+16..base+23
    acc = __builtin_amdgcn_wmma_f32_16x16x32_bf16(false, a.v, false, b.v,
                                                  (short)0, acc, false, false);
  }
  int col = tileN * 16 + l16;
  float bcol = bias[col];
  #pragma unroll
  for (int i = 0; i < 8; ++i) {
    int m = tileM * 16 + i + half * 8;
    float v = acc[i] + bcol;
    if (MODE == 0) {
      ((unsigned short*)outp)[(size_t)m * N + col] = f2bf(v);
    } else if (MODE == 1) {
      int tok = winmap(m);  // window reverse + un-shift scatter
      ((float*)outp)[(size_t)tok * CDIM + col] =
          resid[(size_t)tok * CDIM + col] + v;
    } else if (MODE == 2) {
      float gl = 0.5f * v * (1.0f + erff(v * 0.7071067811865475f));
      ((unsigned short*)outp)[(size_t)m * N + col] = f2bf(gl);
    } else {
      ((float*)outp)[(size_t)m * N + col] = resid[(size_t)m * N + col] + v;
    }
  }
}

// ---------- windowed attention: one block per (window, head) ----------
__global__ __launch_bounds__(128)
void attn_kernel(const unsigned short* __restrict__ qkv,   // (NTOK x 1152) bf16, window order
                 const float* __restrict__ bm,             // (64,12,64,64) bias+mask
                 unsigned short* __restrict__ aout) {      // (NTOK x 384) bf16
  const int h   = blockIdx.x;          // head
  const int gw  = blockIdx.y;          // global window 0..2047
  const int w64 = gw & 63;
  const int tid  = threadIdx.x;
  const int wave = tid >> 5;
  const int lane = tid & 31;
  const int l16  = lane & 15;
  const int half = lane >> 4;

  __shared__ float          S[64 * 65];        // scores, padded stride
  __shared__ unsigned short P[64 * 72];        // softmaxed probs, bf16
  __shared__ unsigned short Vb[64 * 40];       // V tile, zero padded rows

  const float* bmw = bm + (((size_t)w64 * 12 + h) << 12);

  // stage V (49x32) into LDS
  for (int idx = tid; idx < 64 * 16; idx += 128) {
    int r = idx >> 4, cp = (idx & 15) << 1;
    unsigned v = 0;
    if (r < 49)
      v = *(const unsigned*)(qkv + (size_t)(gw * 49 + r) * 1152 + 768 + h * 32 + cp);
    *(unsigned*)(Vb + r * 40 + cp) = v;
  }

  const u32x4 z4 = {0, 0, 0, 0};
  const float scale = 0.17677669529663687f;    // 1/sqrt(32)

  // A fragment: q rows for this wave's M-tile (K = 32 = head_dim)
  int mrow = wave * 16 + l16;
  int qr   = mrow < 49 ? mrow : 48;
  const unsigned short* qbase = qkv + (size_t)(gw * 49 + qr) * 1152 + h * 32;
  Frag a;
  a.q[0] = *(const u32x4*)(qbase + half * 8);
  a.q[1] = *(const u32x4*)(qbase + half * 8 + 16);
  if (mrow >= 49) { a.q[0] = z4; a.q[1] = z4; }

  // S = q k^T * scale + (bias+mask) table; padded cols come out ~-1e30
  #pragma unroll
  for (int nt = 0; nt < 4; ++nt) {
    int nrow = nt * 16 + l16;
    int kr   = nrow < 49 ? nrow : 48;
    const unsigned short* kbase = qkv + (size_t)(gw * 49 + kr) * 1152 + 384 + h * 32;
    Frag bfr;
    bfr.q[0] = *(const u32x4*)(kbase + half * 8);
    bfr.q[1] = *(const u32x4*)(kbase + half * 8 + 16);
    if (nrow >= 49) { bfr.q[0] = z4; bfr.q[1] = z4; }
    v8f acc = {};
    acc = __builtin_amdgcn_wmma_f32_16x16x32_bf16(false, a.v, false, bfr.v,
                                                  (short)0, acc, false, false);
    #pragma unroll
    for (int i = 0; i < 8; ++i) {
      int m = wave * 16 + i + half * 8;
      int n = nt * 16 + l16;
      S[m * 65 + n] = fmaf(acc[i], scale, bmw[(m << 6) + n]);
    }
  }
  __syncthreads();

  // row softmax (one thread per row)
  if (tid < 64) {
    float mx = -1e30f;
    for (int c = 0; c < 64; ++c) mx = fmaxf(mx, S[tid * 65 + c]);
    float sum = 0.f;
    for (int c = 0; c < 64; ++c) {
      float e = __expf(S[tid * 65 + c] - mx);
      S[tid * 65 + c] = e; sum += e;
    }
    float inv = 1.f / sum;
    for (int c = 0; c < 64; ++c)
      P[tid * 72 + c] = f2bf(S[tid * 65 + c] * inv);
  }
  __syncthreads();

  // O = P(49x64pad) @ V(64x32): each wave does its M-tile x two N-tiles
  #pragma unroll
  for (int nt = 0; nt < 2; ++nt) {
    v8f acc = {};
    #pragma unroll
    for (int kk = 0; kk < 64; kk += 32) {
      Frag pa, vb;
      const unsigned short* prow = P + (wave * 16 + l16) * 72 + kk + half * 8;
      pa.q[0] = *(const u32x4*)(prow);
      pa.q[1] = *(const u32x4*)(prow + 16);
      int n = nt * 16 + l16;
      #pragma unroll
      for (int j = 0; j < 16; ++j)
        vb.h[j] = Vb[(kk + half * 8 + (j & 7) + ((j >> 3) << 4)) * 40 + n];
      acc = __builtin_amdgcn_wmma_f32_16x16x32_bf16(false, pa.v, false, vb.v,
                                                    (short)0, acc, false, false);
    }
    #pragma unroll
    for (int i = 0; i < 8; ++i) {
      int m = wave * 16 + i + half * 8;
      if (m < 49)
        aout[(size_t)(gw * 49 + m) * 384 + h * 32 + nt * 16 + l16] = f2bf(acc[i]);
    }
  }
}

// ---------- launch ----------
extern "C" void kernel_launch(void* const* d_in, const int* in_sizes, int n_in,
                              void* d_out, int out_size, void* d_ws, size_t ws_size,
                              hipStream_t stream) {
  (void)in_sizes; (void)n_in; (void)out_size; (void)ws_size;
  const float* hidden = (const float*)d_in[0];
  const float* ln1_g  = (const float*)d_in[1];
  const float* ln1_b  = (const float*)d_in[2];
  const float* qkv_w  = (const float*)d_in[3];
  const float* qkv_b  = (const float*)d_in[4];
  const float* out_w  = (const float*)d_in[5];
  const float* out_b  = (const float*)d_in[6];
  const float* btab   = (const float*)d_in[7];
  const float* ln2_g  = (const float*)d_in[8];
  const float* ln2_b  = (const float*)d_in[9];
  const float* fc1_w  = (const float*)d_in[10];
  const float* fc1_b  = (const float*)d_in[11];
  const float* fc2_w  = (const float*)d_in[12];
  const float* fc2_b  = (const float*)d_in[13];

  char* ws = (char*)d_ws;
  size_t off = 0;
  auto take = [&](size_t bytes) {
    char* p = ws + off; off += (bytes + 255) & ~(size_t)255; return p;
  };
  unsigned short* wqkv   = (unsigned short*)take((size_t)1152 * 384 * 2);
  unsigned short* wout   = (unsigned short*)take((size_t)384 * 384 * 2);
  unsigned short* wfc1   = (unsigned short*)take((size_t)1536 * 384 * 2);
  unsigned short* wfc2   = (unsigned short*)take((size_t)384 * 1536 * 2);
  float*          bmtab  = (float*)take((size_t)64 * 12 * 64 * 64 * 4);
  unsigned short* xw_h2  = (unsigned short*)take((size_t)NTOK * 384 * 2);   // xw, later h2
  unsigned short* qkv_a1 = (unsigned short*)take((size_t)NTOK * 1536 * 2);  // qkv, later a1
  unsigned short* attno  = (unsigned short*)take((size_t)NTOK * 384 * 2);
  float*          xbuf   = (float*)take((size_t)NTOK * 384 * 4);

  // bf16 weight copies + bias/mask table
  f32_to_bf16<<<(1152 * 384 + 255) / 256, 256, 0, stream>>>(qkv_w, wqkv, 1152 * 384);
  f32_to_bf16<<<(384 * 384 + 255) / 256, 256, 0, stream>>>(out_w, wout, 384 * 384);
  f32_to_bf16<<<(1536 * 384 + 255) / 256, 256, 0, stream>>>(fc1_w, wfc1, 1536 * 384);
  f32_to_bf16<<<(384 * 1536 + 255) / 256, 256, 0, stream>>>(fc2_w, wfc2, 384 * 1536);
  build_biasmask<<<(64 * 12 * 64 * 64) / 256, 256, 0, stream>>>(btab, bmtab);

  const int mt = NTOK / 16;  // 6272 M-tiles

  // LN1 + shift + window gather -> xw (window-ordered, bf16)
  layernorm_kernel<true><<<NTOK, 128, 0, stream>>>(hidden, ln1_g, ln1_b, xw_h2);
  // QKV projection
  gemm_wmma<0, 384><<<mt * (1152 / 16) / 8, 256, 0, stream>>>(xw_h2, wqkv, qkv_b,
                                                              qkv_a1, nullptr, 1152);
  // windowed attention
  attn_kernel<<<dim3(12, 2048), 128, 0, stream>>>(qkv_a1, bmtab, attno);
  // out projection + window reverse + un-shift + residual -> xbuf (f32)
  gemm_wmma<1, 384><<<mt * (384 / 16) / 8, 256, 0, stream>>>(attno, wout, out_b,
                                                             xbuf, hidden, 384);
  // LN2 -> h2 (bf16, token order)
  layernorm_kernel<false><<<NTOK, 128, 0, stream>>>(xbuf, ln2_g, ln2_b, xw_h2);
  // FC1 + exact GELU -> a1 (bf16)
  gemm_wmma<2, 384><<<mt * (1536 / 16) / 8, 256, 0, stream>>>(xw_h2, wfc1, fc1_b,
                                                              qkv_a1, nullptr, 1536);
  // FC2 + residual -> d_out (f32)
  gemm_wmma<3, 1536><<<mt * (384 / 16) / 8, 256, 0, stream>>>(qkv_a1, wfc2, fc2_b,
                                                              d_out, xbuf, 384);
}
